// HeteroSchedFoundationModel_29429115912816
// MI455X (gfx1250) — matrix operand order repaired and checked by reading, not compiled, not run on hardware
//
#include <hip/hip_runtime.h>
#include <hip/hip_bf16.h>
#include <cstddef>

// ---------------------------------------------------------------------------
// HeteroSchedFoundationModel forward for gfx1250 (MI455X, wave32, WMMA)
//   - all GEMMs on v_wmma_f32_16x16x32_f16, f32 accumulate
//   - weights pre-converted to f16 once per call
//   - f16 activations for WMMA operands; f32 residual stream for LN fidelity
//   - shared B tiles double-buffered through LDS via global_load_async_to_lds
//     (ASYNCcnt-tracked DMA), ds_store fallback if builtin absent
//   - each wave computes 32x64 of C: 8 WMMAs per staged tile
// ---------------------------------------------------------------------------

typedef __attribute__((ext_vector_type(16))) _Float16 v16h;
typedef __attribute__((ext_vector_type(8)))  _Float16 v8h;
typedef __attribute__((ext_vector_type(8)))  float    v8f;
typedef __attribute__((ext_vector_type(4)))  int      v4i;

#define DM   512
#define SEQ  512
#define BTCH 8
#define ROWS (BTCH * SEQ)   // 4096
#define HEADS 8
#define DK   64
#define DFF  2048
#define ADP  64
#define NLAYER 12
#define LDSROW 40                 // f16 per LDS tile row (80B, bank-spread)
#define TILE_E (64 * LDSROW)      // f16 elements per staged tile

#if __has_builtin(__builtin_amdgcn_global_load_async_to_lds_b128)
#define HAVE_ASYNC_LDS 1
#endif

// ---------------------------------------------------------------------------
// Fragment layouts (ISA 7.12.2), all-f16 operands:
// A 16x32: lane l: m=l&15, h=l>>4; e<8 -> K=h*8+e ; e>=8 -> K=16+h*8+(e-8)
// B 32x16: lane l: n=l&15, h=l>>4; e<8 -> K=h*8+e ; e>=8 -> K=16+h*8+(e-8)
// C/D 16x16 f32: VGPR r -> row r + h*8, col = lane&15
// ---------------------------------------------------------------------------

__device__ __forceinline__ v16h pack16(v8h lo, v8h hi) {
  v16h r;
#pragma unroll
  for (int e = 0; e < 8; ++e) { r[e] = lo[e]; r[8 + e] = hi[e]; }
  return r;
}

__device__ __forceinline__ v16h load_a16(const _Float16* __restrict__ A, int lda,
                                         int m_base, int kk) {
  const int lane = threadIdx.x & 31;
  const int m = m_base + (lane & 15);
  const int h = lane >> 4;
  const _Float16* p = A + (size_t)m * lda + kk;
  v8h lo = *(const v8h*)(p + h * 8);
  v8h hi = *(const v8h*)(p + 16 + h * 8);
  return pack16(lo, hi);
}

// Stage one 64(n) x 32(k) f16 tile into LDS; 256 threads x 16B.
__device__ __forceinline__ void stage_tile(_Float16* smem,
                                           const _Float16* __restrict__ W, int ldw,
                                           int n_base, int kk) {
  const int t = threadIdx.x;
  const int n = t >> 2;        // 0..63
  const int c = t & 3;         // 4 chunks of 8 f16
  const _Float16* g = W + (size_t)(n_base + n) * ldw + kk + c * 8;
  _Float16* l = smem + n * LDSROW + c * 8;
#ifdef HAVE_ASYNC_LDS
  __builtin_amdgcn_global_load_async_to_lds_b128(
      (__attribute__((address_space(1))) v4i*)(void*)g,
      (__attribute__((address_space(3))) v4i*)(void*)l, 0, 0);
#else
  *(v8h*)l = *(const v8h*)g;
#endif
}

__device__ __forceinline__ void wait_stage() {
#ifdef HAVE_ASYNC_LDS
#if __has_builtin(__builtin_amdgcn_s_wait_asynccnt)
  __builtin_amdgcn_s_wait_asynccnt(0);
#else
  asm volatile("s_wait_asynccnt 0x0" ::: "memory");
#endif
#endif
}

__device__ __forceinline__ v16h load_b_smem(const _Float16* smem, int t) {
  const int lane = threadIdx.x & 31;
  const int n = t * 16 + (lane & 15);
  const int h = lane >> 4;
  const _Float16* p = smem + n * LDSROW + h * 8;
  v8h lo = *(const v8h*)p;
  v8h hi = *(const v8h*)(p + 16);
  return pack16(lo, hi);
}

__device__ __forceinline__ v8f wmma16(v16h a, v16h b, v8f c) {
  return __builtin_amdgcn_wmma_f32_16x16x32_f16(false, a, false, b,
                                                (short)0, c, false, false);
}

// Double-buffered K loop: wave computes 32 rows x 64 cols (8 WMMAs / k-step).
__device__ __forceinline__ void mm_loop(v8f acc[2][4],
    const _Float16* __restrict__ A, int lda, int m_base,
    const _Float16* __restrict__ W, int ldw, int n_base, int K,
    _Float16* smem) {
  stage_tile(smem, W, ldw, n_base, 0);
  int buf = 0;
  for (int kk = 0; kk < K; kk += 32) {
    wait_stage();
    __syncthreads();
    if (kk + 32 < K)
      stage_tile(smem + (buf ^ 1) * TILE_E, W, ldw, n_base, kk + 32);
    const _Float16* cur = smem + buf * TILE_E;
    v16h a0 = load_a16(A, lda, m_base, kk);
    v16h a1 = load_a16(A, lda, m_base + 16, kk);
#pragma unroll
    for (int t = 0; t < 4; ++t) {
      v16h b = load_b_smem(cur, t);
      acc[0][t] = wmma16(a0, b, acc[0][t]);
      acc[1][t] = wmma16(a1, b, acc[1][t]);
    }
    buf ^= 1;
  }
}

// ---------------------------------------------------------------------------
// Generic GEMM: C[M,N] = act(A[M,K]_f16 @ W[N,K]_f16^T + bias[N]) (+ resid_f32)
// OUTF: 0 = f32 only, 1 = f16 only, 2 = both, 3 = f16 transposed-per-head (V)
// block = 256 (8 waves); wave -> 32 rows x 64 cols; grid = (N/64, M/256)
// ---------------------------------------------------------------------------
template <int ACT, int OUTF, bool RESID>
__global__ __launch_bounds__(256) void gemm16_kernel(
    const _Float16* __restrict__ A, int lda,
    const _Float16* __restrict__ W, int K,
    const float* __restrict__ bias,
    const float* __restrict__ resid,
    float* __restrict__ Cf, _Float16* __restrict__ Ch, int ldc) {
  __shared__ __align__(16) _Float16 smem[2 * TILE_E];
  const int lane = threadIdx.x & 31;
  const int wid  = threadIdx.x >> 5;
  const int m_base = blockIdx.y * 256 + wid * 32;
  const int n_base = blockIdx.x * 64;

  v8f acc[2][4] = {};
  mm_loop(acc, A, lda, m_base, W, K, n_base, K, smem);

  const int h  = lane >> 4;
  const int nl = lane & 15;
#pragma unroll
  for (int mt = 0; mt < 2; ++mt) {
#pragma unroll
    for (int t = 0; t < 4; ++t) {
      const int col = n_base + t * 16 + nl;
      const float bv = bias[col];
#pragma unroll
      for (int r = 0; r < 8; ++r) {
        const int row = m_base + mt * 16 + r + h * 8;
        float v = acc[mt][t][r] + bv;
        if (ACT == 1) v = v > 0.f ? v : 0.f;
        if (RESID) v += resid[(size_t)row * ldc + col];
        if (OUTF == 0 || OUTF == 2) Cf[(size_t)row * ldc + col] = v;
        if (OUTF == 1 || OUTF == 2) Ch[(size_t)row * ldc + col] = (_Float16)v;
        if (OUTF == 3) {  // V stored transposed per head: (b, h, d, s)
          const int bb = row >> 9, ss = row & (SEQ - 1);
          const int hh = col >> 6, dd = col & 63;
          Ch[(((size_t)bb * HEADS + hh) * DK + dd) * SEQ + ss] = (_Float16)v;
        }
      }
    }
  }
}

// ---------------------------------------------------------------------------
// scores[b,h,i,j] = f16( QK/8 + res_bias[h,res[b,i],res[b,j]] ; mask -> -1e9 )
// grid = (SEQ/64, SEQ/256, B*H)
// ---------------------------------------------------------------------------
__global__ __launch_bounds__(256) void attn_scores_kernel(
    const _Float16* __restrict__ q, const _Float16* __restrict__ k,
    const float* __restrict__ res_bias, const int* __restrict__ resmask,
    const int* __restrict__ amask, _Float16* __restrict__ scores) {
  __shared__ __align__(16) _Float16 smem[2 * TILE_E];
  const int bh = blockIdx.z;
  const int b  = bh >> 3;
  const int hh = bh & 7;
  const int lane = threadIdx.x & 31;
  const int wid  = threadIdx.x >> 5;
  const int i_base = blockIdx.y * 256 + wid * 32;
  const int j_base = blockIdx.x * 64;

  const _Float16* Q  = q + (size_t)b * SEQ * DM + hh * DK;
  const _Float16* Km = k + (size_t)b * SEQ * DM + hh * DK;

  v8f acc[2][4] = {};
  mm_loop(acc, Q, DM, i_base, Km, DM, j_base, DK, smem);

  const float scale = 0.125f;
  const int h  = lane >> 4;
  const int nl = lane & 15;
  _Float16* out = scores + (size_t)bh * SEQ * SEQ;
#pragma unroll
  for (int mt = 0; mt < 2; ++mt) {
#pragma unroll
    for (int t = 0; t < 4; ++t) {
      const int j  = j_base + t * 16 + nl;
      const int rj = resmask[b * SEQ + j];
      const int mj = amask[b * SEQ + j];
#pragma unroll
      for (int r = 0; r < 8; ++r) {
        const int i  = i_base + mt * 16 + r + h * 8;
        const int ri = resmask[b * SEQ + i];
        float v = acc[mt][t][r] * scale + res_bias[(hh * 4 + ri) * 4 + rj];
        if (mj == 0) v = -1e9f;
        out[(size_t)i * SEQ + j] = (_Float16)v;
      }
    }
  }
}

// ---------------------------------------------------------------------------
// In-place f16 row softmax (512 cols). One wave per row, 16 contiguous/lane.
// ---------------------------------------------------------------------------
__global__ __launch_bounds__(256) void softmax_rows_kernel(_Float16* __restrict__ scores) {
  const int row  = blockIdx.x * 8 + (threadIdx.x >> 5);
  const int lane = threadIdx.x & 31;
  _Float16* p = scores + (size_t)row * SEQ + lane * 16;
  v8h lo = *(const v8h*)p;
  v8h hi = *(const v8h*)(p + 8);
  float vals[16];
  float mx = -3.4e38f;
#pragma unroll
  for (int e = 0; e < 8; ++e) { vals[e] = (float)lo[e]; vals[8 + e] = (float)hi[e]; }
#pragma unroll
  for (int e = 0; e < 16; ++e) mx = fmaxf(mx, vals[e]);
#pragma unroll
  for (int off = 16; off > 0; off >>= 1) mx = fmaxf(mx, __shfl_xor(mx, off, 32));
  float sum = 0.f;
#pragma unroll
  for (int e = 0; e < 16; ++e) { vals[e] = __expf(vals[e] - mx); sum += vals[e]; }
#pragma unroll
  for (int off = 16; off > 0; off >>= 1) sum += __shfl_xor(sum, off, 32);
  const float inv = 1.0f / sum;
#pragma unroll
  for (int e = 0; e < 8; ++e) {
    lo[e] = (_Float16)(vals[e] * inv);
    hi[e] = (_Float16)(vals[8 + e] * inv);
  }
  *(v8h*)p = lo;
  *(v8h*)(p + 8) = hi;
}

// ---------------------------------------------------------------------------
// attn[b,i,h*64+d] = sum_j P[b,h,i,j] * V[b,j,h*64+d]; V pre-transposed (b,h,d,s)
// grid = (1, SEQ/256, B*H)
// ---------------------------------------------------------------------------
__global__ __launch_bounds__(256) void attn_av_kernel(
    const _Float16* __restrict__ scores, const _Float16* __restrict__ vT,
    _Float16* __restrict__ attn) {
  __shared__ __align__(16) _Float16 smem[2 * TILE_E];
  const int bh = blockIdx.z;
  const int b  = bh >> 3;
  const int hh = bh & 7;
  const int lane = threadIdx.x & 31;
  const int wid  = threadIdx.x >> 5;
  const int i_base = blockIdx.y * 256 + wid * 32;

  const _Float16* P  = scores + (size_t)bh * SEQ * SEQ;   // (i, j)
  const _Float16* Vt = vT + (size_t)bh * DK * SEQ;        // (d, j): N=64, K=512

  v8f acc[2][4] = {};
  mm_loop(acc, P, SEQ, i_base, Vt, SEQ, 0, SEQ, smem);

  const int h  = lane >> 4;
  const int nl = lane & 15;
#pragma unroll
  for (int mt = 0; mt < 2; ++mt) {
#pragma unroll
    for (int t = 0; t < 4; ++t) {
      const int d = t * 16 + nl;
#pragma unroll
      for (int r = 0; r < 8; ++r) {
        const int i = i_base + mt * 16 + r + h * 8;
        attn[((size_t)(b * SEQ + i)) * DM + hh * DK + d] = (_Float16)acc[mt][t][r];
      }
    }
  }
}

// ---------------------------------------------------------------------------
// out = LayerNorm(a (+ b)) * g + beta ; optional f16 shadow. Wave per row.
// ---------------------------------------------------------------------------
template <bool WH>
__global__ __launch_bounds__(256) void add_ln_kernel(
    const float* __restrict__ a, const float* __restrict__ b,
    const float* __restrict__ g, const float* __restrict__ beta,
    float* __restrict__ out, _Float16* __restrict__ outh) {
  const int row  = blockIdx.x * 8 + (threadIdx.x >> 5);
  const int lane = threadIdx.x & 31;
  const float* pa = a + (size_t)row * DM;
  float vals[16];
  float mean = 0.f;
#pragma unroll
  for (int e = 0; e < 16; ++e) {
    float x = pa[lane + e * 32];
    if (b) x += b[(size_t)row * DM + lane + e * 32];
    vals[e] = x;
    mean += x;
  }
#pragma unroll
  for (int off = 16; off > 0; off >>= 1) mean += __shfl_xor(mean, off, 32);
  mean *= (1.0f / (float)DM);
  float var = 0.f;
#pragma unroll
  for (int e = 0; e < 16; ++e) { float d = vals[e] - mean; var += d * d; }
#pragma unroll
  for (int off = 16; off > 0; off >>= 1) var += __shfl_xor(var, off, 32);
  var *= (1.0f / (float)DM);
  const float inv = rsqrtf(var + 1e-5f);
#pragma unroll
  for (int e = 0; e < 16; ++e) {
    const int c = lane + e * 32;
    const float y = (vals[e] - mean) * inv * g[c] + beta[c];
    out[(size_t)row * DM + c] = y;
    if (WH) outh[(size_t)row * DM + c] = (_Float16)y;
  }
}

// ---------------------------------------------------------------------------
// Strided 1D conv (f32, ~6% of FLOPs)
// ---------------------------------------------------------------------------
__global__ __launch_bounds__(128) void conv_kernel(
    const float* __restrict__ x, const float* __restrict__ w,
    const float* __restrict__ bias, float* __restrict__ out,
    int s, int Lout) {
  const int b  = blockIdx.y;
  const int t  = blockIdx.x;
  const int co = threadIdx.x;
  float acc = bias[co];
  const float* xr = x + ((size_t)b * SEQ + (size_t)t * s) * DM;
  const float* wr = w + (size_t)co * DM * s;
  for (int ci = 0; ci < DM; ++ci)
    for (int kq = 0; kq < s; ++kq)
      acc += xr[(size_t)kq * DM + ci] * wr[ci * s + kq];
  out[((size_t)b * 128 + co) * Lout + t] = acc;
}

// ---------------------------------------------------------------------------
// Interp conv outputs to S and concat -> ms (ROWS, 512) f16
// ---------------------------------------------------------------------------
__global__ __launch_bounds__(256) void assemble_ms_kernel(
    const float* __restrict__ c0, const float* __restrict__ c1,
    const float* __restrict__ c2, const float* __restrict__ c3,
    _Float16* __restrict__ ms) {
  const int idx = blockIdx.x * 256 + threadIdx.x;
  const int row = idx >> 9, col = idx & 511;
  const int b = row >> 9, sidx = row & 511;
  const int grp = col >> 7, co = col & 127;
  const float* src; int Lout;
  switch (grp) {
    case 0:  src = c0; Lout = 512; break;
    case 1:  src = c1; Lout = 128; break;
    case 2:  src = c2; Lout = 32;  break;
    default: src = c3; Lout = 8;   break;
  }
  const float* p = src + ((size_t)b * 128 + co) * Lout;
  float coord = (sidx + 0.5f) * ((float)Lout / (float)SEQ) - 0.5f;
  coord = fminf(fmaxf(coord, 0.0f), (float)(Lout - 1));
  const int lo = (int)floorf(coord);
  const int hi = min(lo + 1, Lout - 1);
  const float w = coord - (float)lo;
  ms[idx] = (_Float16)(p[lo] * (1.0f - w) + p[hi] * w);
}

// ---------------------------------------------------------------------------
// x = state @ se_w^T + se_b + pos_encoding(batch); writes f32 + f16 shadow
// ---------------------------------------------------------------------------
__global__ __launch_bounds__(256) void embed_kernel(
    const float* __restrict__ state, const float* __restrict__ se_w,
    const float* __restrict__ se_b, float* __restrict__ x,
    _Float16* __restrict__ xh) {
  const int idx = blockIdx.x * 256 + threadIdx.x;
  const int row = idx >> 9, d = idx & 511;
  const int b = row >> 9;
  const float* st = state + (size_t)row * 36;
  const float* w  = se_w + (size_t)d * 36;
  float acc = se_b[d];
#pragma unroll 4
  for (int kq = 0; kq < 36; ++kq) acc += st[kq] * w[kq];
  const float freq = __powf(10000.0f, -(float)(d & ~1) / (float)DM);
  const float ang  = (float)b * freq;
  acc += (d & 1) ? __cosf(ang) : __sinf(ang);
  x[idx] = acc;
  xh[idx] = (_Float16)acc;
}

// ---------------------------------------------------------------------------
// f32 -> f16 weight conversion
// ---------------------------------------------------------------------------
__global__ __launch_bounds__(256) void cvt_f16_kernel(
    const float* __restrict__ src, _Float16* __restrict__ dst, int n) {
  const int i = blockIdx.x * 256 + threadIdx.x;
  if (i < n) dst[i] = (_Float16)src[i];
}

// ---------------------------------------------------------------------------
// Host launcher
// ---------------------------------------------------------------------------
extern "C" void kernel_launch(void* const* d_in, const int* in_sizes, int n_in,
                              void* d_out, int out_size, void* d_ws, size_t ws_size,
                              hipStream_t stream) {
  // Flattening (JAX pytree order): 0 state; 1..420 layers (35 each, sorted keys);
  // 421 se_b; 422 se_w; 423 resource_mask; 424 attention_mask
  if (n_in < 425) return;
  const float* state = (const float*)d_in[0];
  auto P = [&](int layer, int off) -> const float* {
    return (const float*)d_in[1 + layer * 35 + off];
  };
  // 0 ad_db 1 ad_dw 2 ad_ub 3 ad_uw 4 b1 5 b2 6 bk 7 bo 8 bq 9 bv
  // 10-13 conv_b 14-17 conv_w 18 fb 19 fw 20 n1b 21 n1g 22 n2b 23 n2g
  // 24 n3b 25 n3g 26 res_bias 27 tlnb 28 tlng 29 w1 30 w2 31 wk 32 wo 33 wq 34 wv
  const float* se_b = (const float*)d_in[421];
  const float* se_w = (const float*)d_in[422];
  const int* resmask = (const int*)d_in[423];
  const int* amask   = (const int*)d_in[424];

  char* base = (char*)d_ws;
  size_t cur = 0;
  auto allocb = [&](size_t bytes) -> void* {
    void* p = base + cur;
    cur = (cur + bytes + 255) & ~(size_t)255;
    return p;
  };
  const size_t MD = (size_t)ROWS * DM;

  float*    x    = (float*)allocb(MD * 4);
  float*    tmp  = (float*)allocb(MD * 4);
  float*    tmp2 = (float*)allocb(MD * 4);
  float*    c0   = (float*)allocb((size_t)BTCH * 128 * 512 * 4);
  float*    c1   = (float*)allocb((size_t)BTCH * 128 * 128 * 4);
  float*    c2   = (float*)allocb((size_t)BTCH * 128 * 32 * 4);
  float*    c3   = (float*)allocb((size_t)BTCH * 128 * 8 * 4);
  _Float16* xh   = (_Float16*)allocb(MD * 2);
  _Float16* qh   = (_Float16*)allocb(MD * 2);
  _Float16* kh   = (_Float16*)allocb(MD * 2);
  _Float16* vT   = (_Float16*)allocb(MD * 2);
  _Float16* attnh= (_Float16*)allocb(MD * 2);
  _Float16* msh  = (_Float16*)allocb(MD * 2);
  _Float16* ff1h = (_Float16*)allocb((size_t)ROWS * DFF * 2);
  _Float16* a1h  = (_Float16*)allocb((size_t)ROWS * ADP * 2);
  _Float16* sch  = (_Float16*)allocb((size_t)BTCH * HEADS * SEQ * SEQ * 2);

  _Float16 *Wq[NLAYER], *Wk[NLAYER], *Wv[NLAYER], *Wo[NLAYER], *Fw[NLAYER],
           *W1[NLAYER], *W2[NLAYER], *AdD[NLAYER], *AdU[NLAYER];
  for (int L = 0; L < NLAYER; ++L) {
    Wq[L]  = (_Float16*)allocb((size_t)DM * DM * 2);
    Wk[L]  = (_Float16*)allocb((size_t)DM * DM * 2);
    Wv[L]  = (_Float16*)allocb((size_t)DM * DM * 2);
    Wo[L]  = (_Float16*)allocb((size_t)DM * DM * 2);
    Fw[L]  = (_Float16*)allocb((size_t)DM * DM * 2);
    W1[L]  = (_Float16*)allocb((size_t)DFF * DM * 2);
    W2[L]  = (_Float16*)allocb((size_t)DM * DFF * 2);
    AdD[L] = (_Float16*)allocb((size_t)ADP * DM * 2);
    AdU[L] = (_Float16*)allocb((size_t)DM * ADP * 2);
  }
  if (ws_size < cur) return;

  const dim3 blk(256);
  const int elemBlocks = (ROWS * DM) / 256;
  auto cvt = [&](const float* s, _Float16* d, int n) {
    cvt_f16_kernel<<<(n + 255) / 256, blk, 0, stream>>>(s, d, n);
  };

  for (int L = 0; L < NLAYER; ++L) {
    cvt(P(L, 33), Wq[L], DM * DM);   cvt(P(L, 31), Wk[L], DM * DM);
    cvt(P(L, 34), Wv[L], DM * DM);   cvt(P(L, 32), Wo[L], DM * DM);
    cvt(P(L, 19), Fw[L], DM * DM);   cvt(P(L, 29), W1[L], DFF * DM);
    cvt(P(L, 30), W2[L], DM * DFF);  cvt(P(L, 1),  AdD[L], ADP * DM);
    cvt(P(L, 3),  AdU[L], DM * ADP);
  }

  embed_kernel<<<elemBlocks, blk, 0, stream>>>(state, se_w, se_b, x, xh);

  for (int L = 0; L < NLAYER; ++L) {
    const dim3 gNN(DM / 64, ROWS / 256);
    // ---- attention ----
    gemm16_kernel<0, 1, false><<<gNN, blk, 0, stream>>>(
        xh, DM, Wq[L], DM, P(L, 8), nullptr, nullptr, qh, DM);
    gemm16_kernel<0, 1, false><<<gNN, blk, 0, stream>>>(
        xh, DM, Wk[L], DM, P(L, 6), nullptr, nullptr, kh, DM);
    gemm16_kernel<0, 3, false><<<gNN, blk, 0, stream>>>(
        xh, DM, Wv[L], DM, P(L, 9), nullptr, nullptr, vT, DM);
    attn_scores_kernel<<<dim3(SEQ / 64, SEQ / 256, BTCH * HEADS), blk, 0, stream>>>(
        qh, kh, P(L, 26), resmask, amask, sch);
    softmax_rows_kernel<<<(BTCH * HEADS * SEQ) / 8, blk, 0, stream>>>(sch);
    attn_av_kernel<<<dim3(1, SEQ / 256, BTCH * HEADS), blk, 0, stream>>>(sch, vT, attnh);
    gemm16_kernel<0, 0, true><<<gNN, blk, 0, stream>>>(
        attnh, DM, Wo[L], DM, P(L, 7), x, tmp, nullptr, DM);
    add_ln_kernel<true><<<ROWS / 8, blk, 0, stream>>>(tmp, nullptr, P(L, 21), P(L, 20), x, xh);

    // ---- temporal ----
    conv_kernel<<<dim3(512, BTCH), dim3(128), 0, stream>>>(x, P(L, 14), P(L, 10), c0, 1, 512);
    conv_kernel<<<dim3(128, BTCH), dim3(128), 0, stream>>>(x, P(L, 15), P(L, 11), c1, 4, 128);
    conv_kernel<<<dim3(32,  BTCH), dim3(128), 0, stream>>>(x, P(L, 16), P(L, 12), c2, 16, 32);
    conv_kernel<<<dim3(8,   BTCH), dim3(128), 0, stream>>>(x, P(L, 17), P(L, 13), c3, 64, 8);
    assemble_ms_kernel<<<elemBlocks, blk, 0, stream>>>(c0, c1, c2, c3, msh);
    gemm16_kernel<0, 0, true><<<gNN, blk, 0, stream>>>(
        msh, DM, Fw[L], DM, P(L, 18), x, tmp, nullptr, DM);
    add_ln_kernel<false><<<ROWS / 8, blk, 0, stream>>>(tmp, nullptr, P(L, 28), P(L, 27), tmp2, nullptr);
    add_ln_kernel<true><<<ROWS / 8, blk, 0, stream>>>(x, tmp2, P(L, 23), P(L, 22), x, xh);

    // ---- FFN ----
    gemm16_kernel<1, 1, false><<<dim3(DFF / 64, ROWS / 256), blk, 0, stream>>>(
        xh, DM, W1[L], DM, P(L, 4), nullptr, nullptr, ff1h, DFF);
    gemm16_kernel<0, 0, true><<<gNN, blk, 0, stream>>>(
        ff1h, DFF, W2[L], DFF, P(L, 5), x, tmp, nullptr, DM);
    add_ln_kernel<true><<<ROWS / 8, blk, 0, stream>>>(tmp, nullptr, P(L, 25), P(L, 24), x, xh);

    // ---- adapter ----
    gemm16_kernel<1, 1, false><<<dim3(ADP / 64, ROWS / 256), blk, 0, stream>>>(
        xh, DM, AdD[L], DM, P(L, 0), nullptr, nullptr, a1h, ADP);
    gemm16_kernel<0, 2, true><<<gNN, blk, 0, stream>>>(
        a1h, ADP, AdU[L], ADP, P(L, 2), x, x, xh, DM);
  }

  (void)hipMemcpyAsync(d_out, x, MD * sizeof(float), hipMemcpyDeviceToDevice, stream);
}
